// Generator_7327214207337
// MI455X (gfx1250) — matrix operand order repaired and checked by reading, not compiled, not run on hardware
//
#include <hip/hip_runtime.h>
#include <stdint.h>

// Problem constants (bert-base-chinese shapes from the reference)
#define BB 8
#define TT 128
#define M_TOK 1024        // B*T tokens
#define HH 768            // K (hidden)
#define HU (HH / 2)       // 384 uints (bf16 pairs) per row
#define VV 21128          // N (vocab)
#define KSTEP 32          // K per v_wmma_f32_16x16x32_bf16
#define NKSTEPS (HH / KSTEP)      // 24
#define NSUB 4            // 16-wide N tiles per wave  -> 64 cols/wave
#define MSUB 2            // 16-row  M tiles per wave  -> 32 rows/wave
#define WAVES 8           // waves per block (stacked along M)
#define MBLK (WAVES * 16 * MSUB)  // 256
#define NBLK (NSUB * 16)          // 64

#define SEQ_QUADS (M_TOK * HH / 4)      // 196608 float4s
#define WR_QUADS ((size_t)VV * HH / 4)  // 4056576 float4s

typedef __bf16 bf16_t;
typedef bf16_t v16bf __attribute__((ext_vector_type(16)));
typedef float  v8f   __attribute__((ext_vector_type(8)));
typedef unsigned int u32x4 __attribute__((ext_vector_type(4)));
typedef int i32x4 __attribute__((ext_vector_type(4)));
typedef int i32x8 __attribute__((ext_vector_type(8)));

struct u4x2 { uint4 a, b; };  // 32B carrier for a 16-element bf16 fragment

#if __has_builtin(__builtin_amdgcn_tensor_load_to_lds)
#define HAVE_TDM 1
#else
#define HAVE_TDM 0
#endif

#if __has_builtin(__builtin_amdgcn_s_wait_tensorcnt)
#define WAIT_TENSORCNT(n) __builtin_amdgcn_s_wait_tensorcnt((short)(n))
#else
#define WAIT_TENSORCNT(n) asm volatile("s_wait_tensorcnt %0" ::"i"(n))
#endif

__device__ __forceinline__ unsigned int pack_bf16(float a, float b) {
  // round-to-nearest-even fp32 -> bf16, packed pair (a low 16, b high 16)
  unsigned int ua = __float_as_uint(a);
  unsigned int ub = __float_as_uint(b);
  ua += 0x7FFFu + ((ua >> 16) & 1u);
  ub += 0x7FFFu + ((ub >> 16) & 1u);
  return (ua >> 16) | (ub & 0xFFFF0000u);
}

__device__ __forceinline__ unsigned int float_key(float f) {
  // order-preserving map float -> uint32 (for atomicMax on packed u64)
  unsigned int u = __float_as_uint(f);
  return (u & 0x80000000u) ? ~u : (u | 0x80000000u);
}

// A-fragment from global bf16 (row-major): 16 bf16 contiguous at p[0..7]
__device__ __forceinline__ v16bf load_frag_global(const unsigned int* p) {
  u4x2 r;
  r.a = *(const uint4*)(p);
  r.b = *(const uint4*)(p + 4);
  return __builtin_bit_cast(v16bf, r);
}

// B-fragment from LDS: 16 bf16 contiguous (8 uints)
__device__ __forceinline__ v16bf load_frag_lds(const unsigned int* p) {
  u4x2 r;
  r.a = *(const uint4*)(p);
  r.b = *(const uint4*)(p + 4);
  return __builtin_bit_cast(v16bf, r);
}

#if HAVE_TDM
// Issue one TDM descriptor: copy a (tile0 x 64-row) tile of packed-bf16 Wr
// (8-byte elements) into LDS at ldsOff.  tensor_dim* are the *remaining*
// extent from the tile start so the ragged last N-block zero-fills (ISA ch8:
// OOB reads return zero).  tile0 == 0 => NOP descriptor (pipeline drain).
__device__ __forceinline__ void tdm_load_B(const unsigned int* gsrc, unsigned ldsOff,
                                           int rowsRemain, int t0units, int tile0) {
  unsigned long long ga = (unsigned long long)(size_t)gsrc;
  u32x4 g0;
  g0[0] = 1u;                                   // count=1 valid, no gather
  g0[1] = ldsOff;                               // D#.lds_addr (bytes)
  g0[2] = (unsigned)(ga & 0xFFFFFFFFull);       // global_addr[31:0]
  g0[3] = (unsigned)(ga >> 32) | (2u << 30);    // global_addr[56:32] | type=2
  i32x8 g1;
  g1[0] = 0x00030000;                           // data_size=3 (8 bytes/elem)
  g1[1] = (t0units & 0xFFFF) << 16;             // tensor_dim0[15:0]
  g1[2] = (((unsigned)t0units >> 16) & 0xFFFF)  // tensor_dim0[31:16]
          | ((rowsRemain & 0xFFFF) << 16);      // tensor_dim1[15:0]
  g1[3] = (((unsigned)rowsRemain >> 16) & 0xFFFF)
          | ((tile0 & 0xFFFF) << 16);           // tile_dim0
  g1[4] = 64;                                   // tile_dim1 = 64 rows
  g1[5] = 192;                                  // tensor_dim0_stride (8B units)
  g1[6] = 0;
  g1[7] = 0;
  i32x4 z4 = {0, 0, 0, 0};
#if defined(__clang_major__) && (__clang_major__ >= 23)
  i32x8 z8 = {0, 0, 0, 0, 0, 0, 0, 0};
  __builtin_amdgcn_tensor_load_to_lds(g0, g1, z4, z4, z8, 0);
#else
  __builtin_amdgcn_tensor_load_to_lds(g0, g1, z4, z4, 0);
#endif
}
#endif  // HAVE_TDM

__global__ __launch_bounds__(256) void init_kernel(unsigned long long* __restrict__ best,
                                                   float* __restrict__ out) {
  int i = blockIdx.x * blockDim.x + threadIdx.x;
  if (i < M_TOK) best[i] = 0ull;  // 0 < float_key(x) for any non-NaN x
  if (i == 0) out[0] = 0.0f;
}

// fp32 -> packed bf16 (RNE), 4 elements per thread
__global__ __launch_bounds__(256) void convert_bf16_kernel(const float4* __restrict__ src,
                                                           uint2* __restrict__ dst,
                                                           unsigned int nquads) {
  unsigned int i = blockIdx.x * blockDim.x + threadIdx.x;
  if (i < nquads) {
    float4 f = src[i];
    uint2 o;
    o.x = pack_bf16(f.x, f.y);
    o.y = pack_bf16(f.z, f.w);
    dst[i] = o;
  }
}

// Fused  logits = seq @ Wr^T + br + gumbel ; per-token argmax over V.
// Wave = 32(M) x 64(N) strip (2x4 WMMA accumulators); block = 8 waves along M.
// B tile (64 vocab rows x 32 K bf16 = 4KB) DMAed into double-buffered LDS by
// the Tensor Data Mover (TENSORcnt-tracked), overlapping the WMMAs of the
// previous tile.
__global__ __launch_bounds__(256) void gemm_argmax_kernel(
    const unsigned int* __restrict__ seqBf,  // [M_TOK][HU] packed bf16
    const unsigned int* __restrict__ WrBf,   // [VV][HU]    packed bf16
    const float* __restrict__ br,            // [VV]
    const float* __restrict__ gum,           // [M_TOK][VV]
    unsigned long long* __restrict__ best) { // [M_TOK] packed (key<<32)|~idx
  __shared__ __align__(16) unsigned int ldsB[2 * 64 * 16];  // 2 x 4KB tiles

  const int tid    = threadIdx.x;
  const int lane   = tid & 31;
  const int wave   = tid >> 5;
  const int laneLo = lane & 15;
  const bool hi    = lane >= 16;

  const int mBase = blockIdx.y * MBLK + wave * (16 * MSUB);
  const int nBase = blockIdx.x * NBLK;

  // A fragment bases (16x32 bf16, contiguous k = 16*(lane/16) + s, same order as B):
  // lane row = m-tile base + lane%16; K window base uint = k0/2 + (hi?8:0),
  // halves at +0 and +4 uints.
  const unsigned int* aRow0 = seqBf + (size_t)(mBase + laneLo) * HU + (hi ? 8 : 0);
  const unsigned int* aRow1 = aRow0 + (size_t)16 * HU;

  v8f acc[MSUB][NSUB] = {};

#if HAVE_TDM
  const unsigned int* bTileBase = WrBf + (size_t)nBase * HU;  // row nBase, k=0
  const unsigned ldsOff0 = (unsigned)(size_t)(void*)&ldsB[0]; // low32 = LDS byte addr
  const int rowsRemain = VV - nBase;

  if (wave == 0) tdm_load_B(bTileBase, ldsOff0, rowsRemain, 192, 8);

  for (int ks = 0; ks < NKSTEPS; ++ks) {
    const int ku = ks * (KSTEP / 2);
    if (wave == 0) {
      const int nxt = ks + 1;
      if (nxt < NKSTEPS) {
        tdm_load_B(bTileBase + nxt * 16, ldsOff0 + (unsigned)((nxt & 1) * 4096),
                   rowsRemain, 192 - nxt * 4, 8);
      } else {
        tdm_load_B(bTileBase, ldsOff0, rowsRemain, 192, /*tile0=NOP*/ 0);
      }
      WAIT_TENSORCNT(1);  // in-order per wave: tile `ks` has landed in LDS
    }
    __syncthreads();  // publish tile ks to all 8 waves

    const unsigned int* ldsCur = ldsB + ((ks & 1) ? 1024 : 0);

    v16bf af0 = load_frag_global(aRow0 + ku);
    v16bf af1 = load_frag_global(aRow1 + ku);
#pragma unroll
    for (int t = 0; t < NSUB; ++t) {
      // 32x16 bf16 B layout: lane col = t*16 + lane%16, K half by lane>=16
      v16bf bf = load_frag_lds(&ldsCur[(t * 16 + laneLo) * 16 + (hi ? 8 : 0)]);
      acc[0][t] = __builtin_amdgcn_wmma_f32_16x16x32_bf16(
          false, af0, false, bf, (short)0, acc[0][t], false, false);
      acc[1][t] = __builtin_amdgcn_wmma_f32_16x16x32_bf16(
          false, af1, false, bf, (short)0, acc[1][t], false, false);
    }
    __syncthreads();  // all reads of tile ks done before TDM overwrites it
  }
#else
  // Fallback: synchronous cooperative staging (known-good path)
  const int sr = tid >> 2;
  const int sp = tid & 3;
  int svRow = nBase + sr;
  if (svRow >= VV) svRow = VV - 1;
  const unsigned int* bSrcRow = WrBf + (size_t)svRow * HU + sp * 4;

  for (int k0 = 0; k0 < HH; k0 += KSTEP) {
    const int ku = k0 >> 1;
    uint4 bstage = *(const uint4*)(bSrcRow + ku);
    if (k0 + KSTEP < HH) __builtin_prefetch(bSrcRow + ku + (KSTEP >> 1), 0, 3);
    __syncthreads();
    *(uint4*)&ldsB[sr * 16 + sp * 4] = bstage;
    __syncthreads();

    v16bf af0 = load_frag_global(aRow0 + ku);
    v16bf af1 = load_frag_global(aRow1 + ku);
#pragma unroll
    for (int t = 0; t < NSUB; ++t) {
      v16bf bf = load_frag_lds(&ldsB[(t * 16 + laneLo) * 16 + (hi ? 8 : 0)]);
      acc[0][t] = __builtin_amdgcn_wmma_f32_16x16x32_bf16(
          false, af0, false, bf, (short)0, acc[0][t], false, false);
      acc[1][t] = __builtin_amdgcn_wmma_f32_16x16x32_bf16(
          false, af1, false, bf, (short)0, acc[1][t], false, false);
    }
  }
#endif

  // ---- epilogue: + bias + gumbel, per-row running argmax ----
  // C layout: lane holds col n = tile + lane%16; VGPR j holds row
  // m = tileM + j + (lane>=16 ? 8 : 0).
  float    bv[MSUB][8];
  unsigned bn[MSUB][8];
#pragma unroll
  for (int mi = 0; mi < MSUB; ++mi)
#pragma unroll
    for (int j = 0; j < 8; ++j) { bv[mi][j] = -__builtin_inff(); bn[mi][j] = 0u; }

#pragma unroll
  for (int t = 0; t < NSUB; ++t) {
    int  n     = nBase + t * 16 + laneLo;
    bool valid = (n < VV);
    float bias = valid ? br[n] : 0.0f;
#pragma unroll
    for (int mi = 0; mi < MSUB; ++mi) {
#pragma unroll
      for (int j = 0; j < 8; ++j) {
        float val = -__builtin_inff();
        if (valid) {
          int m = mBase + mi * 16 + j + (hi ? 8 : 0);
          val   = acc[mi][t][j] + bias + gum[(size_t)m * (size_t)VV + (size_t)n];
        }
        if (val > bv[mi][j]) { bv[mi][j] = val; bn[mi][j] = (unsigned)n; }
      }
    }
  }

  // cross-lane reduce within each 16-lane half (ties -> lowest index)
#pragma unroll
  for (int mi = 0; mi < MSUB; ++mi) {
#pragma unroll
    for (int j = 0; j < 8; ++j) {
      float    v = bv[mi][j];
      unsigned n = bn[mi][j];
#pragma unroll
      for (int s = 1; s < 16; s <<= 1) {
        float    ov = __shfl_xor(v, s, 32);
        unsigned on = (unsigned)__shfl_xor((int)n, s, 32);
        if (ov > v || (ov == v && on < n)) { v = ov; n = on; }
      }
      if (laneLo == 0) {
        int m = mBase + mi * 16 + j + (hi ? 8 : 0);
        unsigned long long pk =
            ((unsigned long long)float_key(v) << 32) | (unsigned long long)(~n);
        atomicMax(&best[m], pk);
      }
    }
  }
}

// One block, deterministic tree reduction:
// loss = sum_t chinese[ids[t]] * (1 - sim[ids[t], argmax_t]) / 1024
__global__ __launch_bounds__(1024) void finalize_kernel(
    const unsigned long long* __restrict__ best,
    const int* __restrict__ ids,
    const unsigned char* __restrict__ chmask,  // jax bool -> 1 byte
    const float* __restrict__ sim,             // [VV, VV]
    float* __restrict__ out) {
  __shared__ float red[1024];
  int t = threadIdx.x;
  float c = 0.0f;
  if (t < M_TOK) {
    unsigned long long pk = best[t];
    unsigned idx  = ~(unsigned)(pk & 0xFFFFFFFFull);
    int      orig = ids[t];
    if (chmask[orig] != 0) {
      c = 1.0f - sim[(size_t)orig * (size_t)VV + (size_t)idx];
    }
  }
  red[t] = c;
  __syncthreads();
#pragma unroll
  for (int s = 512; s > 0; s >>= 1) {
    if (t < s) red[t] += red[t + s];
    __syncthreads();
  }
  if (t == 0) out[0] = red[0] * (1.0f / (float)M_TOK);
}

extern "C" void kernel_launch(void* const* d_in, const int* in_sizes, int n_in,
                              void* d_out, int out_size, void* d_ws, size_t ws_size,
                              hipStream_t stream) {
  const float* seq = (const float*)d_in[0];                  // [8,128,768]
  const float* Wr  = (const float*)d_in[1];                  // [21128,768]
  const float* br  = (const float*)d_in[2];                  // [21128]
  const float* sim = (const float*)d_in[3];                  // [21128,21128]
  const float* gum = (const float*)d_in[4];                  // [8,128,21128]
  const int*   ids = (const int*)d_in[5];                    // [8,128]
  const unsigned char* chm = (const unsigned char*)d_in[6];  // [21128] bool
  float* out = (float*)d_out;

  // workspace layout (all 256B-aligned)
  unsigned long long* best = (unsigned long long*)d_ws;                   // 8KB
  unsigned int* seqBf = (unsigned int*)((char*)d_ws + 8192);              // 1.5MB
  unsigned int* WrBf =
      (unsigned int*)((char*)d_ws + 8192 + (size_t)M_TOK * HH * 2);       // 32.5MB

  init_kernel<<<(M_TOK + 255) / 256, 256, 0, stream>>>(best, out);

  convert_bf16_kernel<<<(SEQ_QUADS + 255) / 256, 256, 0, stream>>>(
      (const float4*)seq, (uint2*)seqBf, SEQ_QUADS);
  convert_bf16_kernel<<<(unsigned)((WR_QUADS + 255) / 256), 256, 0, stream>>>(
      (const float4*)Wr, (uint2*)WrBf, (unsigned)WR_QUADS);

  dim3 grid((VV + NBLK - 1) / NBLK, M_TOK / MBLK);  // (331, 4)
  gemm_argmax_kernel<<<grid, 256, 0, stream>>>(seqBf, WrBf, br, gum, best);

  finalize_kernel<<<1, 1024, 0, stream>>>(best, ids, chm, sim, out);
}
